// SlidingKernelAttention_5136780886522
// MI455X (gfx1250) — matrix operand order repaired and checked by reading, not compiled, not run on hardware
//
#include <hip/hip_runtime.h>
#include <hip/hip_fp16.h>
#include <math.h>

typedef __attribute__((ext_vector_type(16))) _Float16 v16h;
typedef __attribute__((ext_vector_type(8)))  _Float16 v8h;
typedef __attribute__((ext_vector_type(4)))  _Float16 v4h;
typedef __attribute__((ext_vector_type(8)))  float    v8f;

#define HEADSq 8
#define KSq 4
#define STq 2
#define Bq 8
#define Lq 2048
#define Cq 1024
#define Dhq 128
#define NWIN 1023            // (L-KS)/ST+1
#define MROWS (Bq*Lq)        // 16384

// ---------------- CDNA5 async global->LDS helpers ----------------

__device__ __forceinline__ void async_b128(void* lds, const void* g) {
  unsigned l = (unsigned)(uintptr_t)lds;                 // low 32b = LDS offset
  unsigned long long a = (unsigned long long)(uintptr_t)g;
  asm volatile("global_load_async_to_lds_b128 %0, %1, off"
               :: "v"(l), "v"(a) : "memory");
}
__device__ __forceinline__ void wait_async4() {
  asm volatile("s_wait_asynccnt 0x4" ::: "memory");
}
__device__ __forceinline__ void wait_async0() {
  asm volatile("s_wait_asynccnt 0x0" ::: "memory");
}

// ---------------- packing kernels ----------------

__global__ void cvt_f32_to_f16(const float* __restrict__ src,
                               _Float16* __restrict__ dst, int n) {
  int i = blockIdx.x * blockDim.x + threadIdx.x;
  int stride = gridDim.x * blockDim.x;
  for (; i < n; i += stride) dst[i] = (_Float16)src[i];
}

// dst[n*K + k] = (half)src[k*N + n]   (weights: make K contiguous per output row)
__global__ void cvt_transpose(const float* __restrict__ src,
                              _Float16* __restrict__ dst, int K, int N) {
  __shared__ float tile[32][33];
  int n0 = blockIdx.x * 32, k0 = blockIdx.y * 32;
  int tx = threadIdx.x & 31, ty = threadIdx.x >> 5;   // 256 thr: ty 0..7
  for (int i = ty; i < 32; i += 8)
    tile[i][tx] = src[(size_t)(k0 + i) * N + n0 + tx];
  __syncthreads();
  for (int i = ty; i < 32; i += 8)
    dst[(size_t)(n0 + i) * K + k0 + tx] = (_Float16)tile[tx][i];
}

__global__ void zero_bytes16(uint4* __restrict__ p, size_t n16) {
  size_t i = (size_t)blockIdx.x * blockDim.x + threadIdx.x;
  size_t stride = (size_t)gridDim.x * blockDim.x;
  uint4 z; z.x = z.y = z.z = z.w = 0u;
  for (; i < n16; i += stride) p[i] = z;
}

// ---------------- WMMA GEMM (async double-buffered) ----------------
// D(MxN) = A(MxK) * B^T where Bm is stored (N,K) row-major, fp16.
// Block tile 128x128, 8 waves, each wave 32(M)x64(N) = 2x4 WMMA 16x16 tiles.
// K staged 32 at a time via GLOBAL_LOAD_ASYNC_TO_LDS_B128 into 2 LDS buffers.
// EPI=0: store fp16 D.  EPI=1: store fp32 D + bias[col]*cnt(row%L).
template <int EPI>
__global__ __launch_bounds__(256) void gemm_wmma(
    const _Float16* __restrict__ A, const _Float16* __restrict__ Bm,
    void* __restrict__ D, int M, int N, int K,
    const float* __restrict__ bias) {
  __shared__ __align__(16) _Float16 As[2][128 * 32];
  __shared__ __align__(16) _Float16 Bs[2][128 * 32];

  const int tid  = threadIdx.x;
  const int lane = tid & 31, wid = tid >> 5;
  const int waveM = wid >> 1;            // 0..3  (32 rows each)
  const int waveN = wid & 1;             // 0..1  (64 cols each)
  const int m0 = blockIdx.y * 128, n0 = blockIdx.x * 128;

  v8f acc[2][4] = {};

  const int lrow = tid >> 1;             // 0..127
  const int lk   = (tid & 1) * 16;       // 0 or 16
  const int r = lane & 15, hh = lane >> 4;

  const _Float16* gA = A  + (size_t)(m0 + lrow) * K + lk;
  const _Float16* gB = Bm + (size_t)(n0 + lrow) * K + lk;

  const int S = K / 32;                  // K steps

  // prologue: stage 0 -> buffer 0 (4 async b128 per thread = 16KB per stage)
  {
    async_b128(&As[0][lrow * 32 + lk],     gA);
    async_b128(&As[0][lrow * 32 + lk + 8], gA + 8);
    async_b128(&Bs[0][lrow * 32 + lk],     gB);
    async_b128(&Bs[0][lrow * 32 + lk + 8], gB + 8);
  }

  for (int i = 0; i < S; ++i) {
    const int cur = i & 1;
    if (i + 1 < S) {
      const int kc = (i + 1) * 32;
      const int nxt = cur ^ 1;
      async_b128(&As[nxt][lrow * 32 + lk],     gA + kc);
      async_b128(&As[nxt][lrow * 32 + lk + 8], gA + kc + 8);
      async_b128(&Bs[nxt][lrow * 32 + lk],     gB + kc);
      async_b128(&Bs[nxt][lrow * 32 + lk + 8], gB + kc + 8);
      wait_async4();                     // in-order: stage i has landed
    } else {
      wait_async0();
    }
    __syncthreads();

    // A fragments: ISA 16-bit A 16x32 layout — per lane two 8-half runs:
    // elem i<8  -> k = hh*8 + i ; elem i>=8 -> k = 16 + hh*8 + (i-8)
    v16h afr[2];
#pragma unroll
    for (int mt = 0; mt < 2; ++mt) {
      const _Float16* p = &As[cur][(waveM * 32 + mt * 16 + r) * 32];
      v8h x0 = *(const v8h*)(p + hh * 8);
      v8h x1 = *(const v8h*)(p + 16 + hh * 8);
      afr[mt] = __builtin_shufflevector(x0, x1, 0, 1, 2, 3, 4, 5, 6, 7,
                                        8, 9, 10, 11, 12, 13, 14, 15);
    }
    // B fragments: lane holds column (lane&15); elem i<8 -> k = hh*8 + i ; elem i>=8 -> k = 16 + hh*8 + (i-8)
    v16h bfr[4];
#pragma unroll
    for (int nt = 0; nt < 4; ++nt) {
      const _Float16* p = &Bs[cur][(waveN * 64 + nt * 16 + r) * 32 + hh * 8];
      v8h x0 = *(const v8h*)p;
      v8h x1 = *(const v8h*)(p + 16);
      bfr[nt] = __builtin_shufflevector(x0, x1, 0, 1, 2, 3, 4, 5, 6, 7,
                                        8, 9, 10, 11, 12, 13, 14, 15);
    }
#pragma unroll
    for (int mt = 0; mt < 2; ++mt)
#pragma unroll
      for (int nt = 0; nt < 4; ++nt)
        acc[mt][nt] = __builtin_amdgcn_wmma_f32_16x16x32_f16(
            false, afr[mt], false, bfr[nt], (short)0, acc[mt][nt], false, false);
    __syncthreads();                     // buf reuse fence for stage i+2 writes
  }

  // epilogue: C/D layout — VGPR e holds row (lane>>4)*8 + e, col lane&15
#pragma unroll
  for (int mt = 0; mt < 2; ++mt)
#pragma unroll
    for (int nt = 0; nt < 4; ++nt) {
      int row = m0 + waveM * 32 + mt * 16 + hh * 8;
      int col = n0 + waveN * 64 + nt * 16 + r;
      if (EPI == 0) {
        _Float16* d = (_Float16*)D;
#pragma unroll
        for (int e = 0; e < 8; ++e)
          d[(size_t)(row + e) * N + col] = (_Float16)acc[mt][nt][e];
      } else {
        float* d = (float*)D;
        float bv = bias[col];
#pragma unroll
        for (int e = 0; e < 8; ++e) {
          int rr = row + e;
          int l = rr & (Lq - 1);
          int j0 = l & 1;
          int c = 0;
          if (((l - j0) >> 1) <= NWIN - 1) c++;
          if (l >= j0 + 2 && ((l - j0 - 2) >> 1) <= NWIN - 1) c++;
          d[(size_t)rr * N + col] = acc[mt][nt][e] + bv * (float)c;
        }
      }
    }
}

// ---------------- windowed attention ----------------
// One block per (b, n) window; each wave handles one head.
// Lanes split Dh=128 into 4-element chunks. Output rows are overlap-added
// into oacc (fp16) with packed-f16 global atomics (position l = 2n+j).
__global__ __launch_bounds__(256) void attn_kernel(
    const _Float16* __restrict__ qkv, _Float16* __restrict__ oacc) {
  int bw = blockIdx.x;              // 0 .. B*NWIN-1
  int b = bw / NWIN, n = bw % NWIN;
  int h = threadIdx.x >> 5, lane = threadIdx.x & 31;
  int d0 = lane * 4;

  const size_t rowbase = ((size_t)b * Lq + (size_t)n * STq) * (3 * Cq);
  const _Float16* qp = qkv + rowbase + (size_t)h * Dhq + d0;
  const _Float16* kp = qp + Cq;
  const _Float16* vp = qp + 2 * Cq;

  float q[KSq][4], k[KSq][4], v[KSq][4];
#pragma unroll
  for (int j = 0; j < KSq; ++j) {
    v4h tq = *(const v4h*)(qp + (size_t)j * 3 * Cq);
    v4h tk = *(const v4h*)(kp + (size_t)j * 3 * Cq);
    v4h tv = *(const v4h*)(vp + (size_t)j * 3 * Cq);
#pragma unroll
    for (int i = 0; i < 4; ++i) {
      q[j][i] = (float)tq[i]; k[j][i] = (float)tk[i]; v[j][i] = (float)tv[i];
    }
  }

  const float scale = 0.03125f;     // C^-0.5 = 1/32
  float attnw[KSq][KSq];
#pragma unroll
  for (int j = 0; j < KSq; ++j) {
    float dots[KSq];
#pragma unroll
    for (int kk = 0; kk < KSq; ++kk) {
      float p = q[j][0] * k[kk][0] + q[j][1] * k[kk][1] +
                q[j][2] * k[kk][2] + q[j][3] * k[kk][3];
#pragma unroll
      for (int off = 16; off > 0; off >>= 1) p += __shfl_xor(p, off, 32);
      dots[kk] = p * scale;
    }
    float m = fmaxf(fmaxf(dots[0], dots[1]), fmaxf(dots[2], dots[3]));
    float e[KSq], s = 0.f;
#pragma unroll
    for (int kk = 0; kk < KSq; ++kk) { e[kk] = __expf(dots[kk] - m); s += e[kk]; }
    float inv = 1.0f / s;
#pragma unroll
    for (int kk = 0; kk < KSq; ++kk) attnw[j][kk] = e[kk] * inv;
  }

#pragma unroll
  for (int j = 0; j < KSq; ++j) {
    float o[4] = {0.f, 0.f, 0.f, 0.f};
#pragma unroll
    for (int kk = 0; kk < KSq; ++kk)
#pragma unroll
      for (int i = 0; i < 4; ++i) o[i] += attnw[j][kk] * v[kk][i];

    size_t l = (size_t)n * STq + j;
    _Float16* dst = oacc + ((size_t)b * Lq + l) * Cq + (size_t)h * Dhq + d0;
#pragma unroll
    for (int p2 = 0; p2 < 2; ++p2) {
      union { _Float16 hv[2]; unsigned int u; } pk;
      pk.hv[0] = (_Float16)o[p2 * 2];
      pk.hv[1] = (_Float16)o[p2 * 2 + 1];
      unsigned long long addr = (unsigned long long)(uintptr_t)(dst + p2 * 2);
      asm volatile("global_atomic_pk_add_f16 %0, %1, off"
                   :: "v"(addr), "v"(pk.u) : "memory");
    }
  }
}

// ---------------- launcher ----------------
extern "C" void kernel_launch(void* const* d_in, const int* in_sizes, int n_in,
                              void* d_out, int out_size, void* d_ws, size_t ws_size,
                              hipStream_t stream) {
  const float* x     = (const float*)d_in[0];
  const float* w_qkv = (const float*)d_in[1];
  const float* w_out = (const float*)d_in[2];
  const float* b_out = (const float*)d_in[3];

  char* ws = (char*)d_ws;
  size_t off = 0;
  auto alloc = [&](size_t bytes) -> void* {
    void* p = ws + off;
    off += (bytes + 255) & ~(size_t)255;
    return p;
  };
  _Float16* xh    = (_Float16*)alloc((size_t)MROWS * Cq * 2);        // 32 MB
  _Float16* wqkvT = (_Float16*)alloc((size_t)3 * Cq * Cq * 2);       // 6 MB
  _Float16* woutT = (_Float16*)alloc((size_t)Cq * Cq * 2);           // 2 MB
  _Float16* qkvh  = (_Float16*)alloc((size_t)MROWS * 3 * Cq * 2);    // 96 MB
  _Float16* oacc  = (_Float16*)alloc((size_t)MROWS * Cq * 2);        // 32 MB

  cvt_f32_to_f16<<<4096, 256, 0, stream>>>(x, xh, MROWS * Cq);
  cvt_transpose<<<dim3(3 * Cq / 32, Cq / 32), 256, 0, stream>>>(w_qkv, wqkvT, Cq, 3 * Cq);
  cvt_transpose<<<dim3(Cq / 32, Cq / 32), 256, 0, stream>>>(w_out, woutT, Cq, Cq);
  zero_bytes16<<<2048, 256, 0, stream>>>((uint4*)oacc, (size_t)MROWS * Cq * 2 / 16);

  // qkv = x @ w_qkv   (fp16 in, fp32 acc, fp16 out)
  gemm_wmma<0><<<dim3(3 * Cq / 128, MROWS / 128), 256, 0, stream>>>(
      xh, wqkvT, qkvh, MROWS, 3 * Cq, Cq, nullptr);

  // windowed attention + overlap-add of o into oacc
  attn_kernel<<<Bq * NWIN, 256, 0, stream>>>(qkvh, oacc);

  // out = oacc @ w_out + cnt(l) * b_out   (fp32 out)
  gemm_wmma<1><<<dim3(Cq / 128, MROWS / 128), 256, 0, stream>>>(
      oacc, woutT, d_out, MROWS, Cq, Cq, b_out);
}
